// MultiModalRetriever_77558519431273
// MI455X (gfx1250) — matrix operand order repaired and checked by reading, not compiled, not run on hardware
//
#include <hip/hip_runtime.h>

// ---------------------------------------------------------------------------
// MultiModalRetriever for MI455X (gfx1250, wave32, WMMA bf16 16x16x32).
// Round 2: branch-free WMMA GEMM with fragment-major ("swizzled") B layout so
// every fragment is 2x global_load_b128 off a single pointer; all edge guards
// removed via M/N padding (padded rows/cols land in scratch no one reads).
// ---------------------------------------------------------------------------

typedef __attribute__((ext_vector_type(16))) __bf16    v16bf;
typedef __attribute__((ext_vector_type(8)))  float     v8f;
typedef __attribute__((ext_vector_type(4)))  unsigned  uint4v;

union ABfrag { v16bf v; unsigned u[8]; uint4v q[2]; };

static __device__ __forceinline__ unsigned short f32_to_bf16(float f) {
  union { float f; unsigned u; } x;
  x.f = f;
  unsigned r = 0x7FFFu + ((x.u >> 16) & 1u);   // round-to-nearest-even
  return (unsigned short)((x.u + r) >> 16);
}

static __device__ __forceinline__ unsigned pack_bf16x2(float lo, float hi) {
  return (unsigned)f32_to_bf16(lo) | ((unsigned)f32_to_bf16(hi) << 16);
}

// ---------------- elementwise f32 -> bf16 ----------------------------------
__global__ void k_f32_to_bf16(const float* __restrict__ x,
                              unsigned short* __restrict__ y, size_t n) {
  size_t i = (size_t)blockIdx.x * blockDim.x + threadIdx.x;
  if (i < n) y[i] = f32_to_bf16(x[i]);
}

// ---- B-tile pack:  X[R x D] f32  ->  Bs[(D/32)][NP][16] dwords ------------
// Bs[(t*NP + n)*16 + s] = { bf16(X[n][t*32+2s]) , bf16(X[n][t*32+2s+1]) }
// (VGPR r of lane half hlf = dword 4*hlf+r (r<4) or 4*hlf+r+4 (r>=4); the hw permutation
//  collapses to plain sequential k-pairs).  Zero-fill for n >= R (padding).
__global__ void k_pack_btile(const float* __restrict__ X,
                             unsigned* __restrict__ Bs, int R, int NP, int D) {
  size_t idx = (size_t)blockIdx.x * blockDim.x + threadIdx.x;
  size_t total = (size_t)(D >> 5) * NP * 16;
  if (idx >= total) return;
  int s = (int)(idx & 15);
  size_t q = idx >> 4;
  int n = (int)(q % NP);
  int t = (int)(q / NP);
  unsigned val = 0u;
  if (n < R) {
    const float* src = X + (size_t)n * D + t * 32 + 2 * s;
    val = pack_bf16x2(src[0], src[1]);
  }
  Bs[idx] = val;
}

// ---------------- branch-free bf16 WMMA GEMM:  C = A * B + bias ------------
// A:  MP x K bf16 row-major (M padded to multiple of 64, pad rows = garbage).
// Bs: (K/32) x NP x 16-dword swizzled tiles (NP multiple of 256).
// C:  MP x NP f32.  Block = 8 waves (2x4), wave tile 32x64, block 64x256.
__global__ __launch_bounds__(256) void k_wmma_gemm(
    const unsigned short* __restrict__ A,
    const unsigned* __restrict__ Bs,
    const float* __restrict__ bias,      // length >= real N or null
    float* __restrict__ C,
    int NP, int K) {
  const int lane = threadIdx.x & 31;
  const int wave = threadIdx.x >> 5;
  const int hlf  = (lane >> 4) & 1;
  const int l15  = lane & 15;
  const int wm = wave >> 2;              // 0..1
  const int wn = wave & 3;               // 0..3
  const int m0 = blockIdx.y * 64 + wm * 32;
  const int n0 = blockIdx.x * 256 + wn * 64;

  // Single base pointer per operand; everything else is immediate offsets.
  const unsigned short* ap = A + (size_t)(m0 + l15) * K + hlf * 8;
  const unsigned* bp = Bs + (size_t)(n0 + l15) * 16 + hlf * 4;

  v8f acc[2][4] = {};

  const int ksteps = K >> 5;
#pragma unroll 4
  for (int t = 0; t < ksteps; ++t) {
    ABfrag a[2];
#pragma unroll
    for (int mi = 0; mi < 2; ++mi) {
      const unsigned short* p = ap + (size_t)mi * 16 * K;
      a[mi].q[0] = *(const uint4v*)(p);
      a[mi].q[1] = *(const uint4v*)(p + 16);
    }
    ABfrag b[4];
#pragma unroll
    for (int ni = 0; ni < 4; ++ni) {
      const unsigned* p = bp + ni * 256;
      b[ni].q[0] = *(const uint4v*)(p);
      b[ni].q[1] = *(const uint4v*)(p + 8);
    }
#pragma unroll
    for (int mi = 0; mi < 2; ++mi)
#pragma unroll
      for (int ni = 0; ni < 4; ++ni)
        acc[mi][ni] = __builtin_amdgcn_wmma_f32_16x16x32_bf16(
            false, a[mi].v, false, b[ni].v, (short)0, acc[mi][ni], false, false);
    ap += 32;
    bp += (size_t)NP * 16;
  }

  // Epilogue: C/D layout: lane col = l15, VGPR r -> row r (+8 hi half).
#pragma unroll
  for (int ni = 0; ni < 4; ++ni) {
    int n = n0 + ni * 16 + l15;
    float bv = bias ? bias[n] : 0.0f;
#pragma unroll
    for (int mi = 0; mi < 2; ++mi) {
#pragma unroll
      for (int r = 0; r < 8; ++r) {
        int m = m0 + mi * 16 + hlf * 8 + r;
        C[(size_t)m * NP + n] = acc[mi][ni][r] + bv;
      }
    }
  }
}

// ---------------- per-head attention scores --------------------------------
// S[(h*32+b)][n] = (1/8) * sum_{k in head h} q[b][k] * kvec[n][k]
// A = q (32 x 512 bf16), Bs = kvec^T swizzled (16 x NCP x 16).  Per wave:
// 16 n-cols, 2 M-tiles x 8 per-head accumulators (128 acc VGPRs).
__global__ __launch_bounds__(256) void k_scores_perhead(
    const unsigned short* __restrict__ Aq,
    const unsigned* __restrict__ Bs,
    float* __restrict__ S, int NCP) {
  const int lane = threadIdx.x & 31;
  const int wave = threadIdx.x >> 5;
  const int hlf  = (lane >> 4) & 1;
  const int l15  = lane & 15;
  const int n = blockIdx.x * 128 + wave * 16 + l15;

  const unsigned short* ap = Aq + (size_t)l15 * 512 + hlf * 8;
  const unsigned* bp = Bs + (size_t)n * 16 + hlf * 4;

  v8f acc0[8] = {};
  v8f acc1[8] = {};

#pragma unroll
  for (int h = 0; h < 8; ++h) {
#pragma unroll
    for (int t2 = 0; t2 < 2; ++t2) {
      const int t = h * 2 + t2;            // k-block index, kk = t*32
      ABfrag a0, a1, b;
      const unsigned short* p0 = ap + t * 32;
      a0.q[0] = *(const uint4v*)(p0);
      a0.q[1] = *(const uint4v*)(p0 + 16);
      const unsigned short* p1 = p0 + 16 * 512;
      a1.q[0] = *(const uint4v*)(p1);
      a1.q[1] = *(const uint4v*)(p1 + 16);
      const unsigned* pb = bp + (size_t)t * NCP * 16;
      b.q[0] = *(const uint4v*)(pb);
      b.q[1] = *(const uint4v*)(pb + 8);
      acc0[h] = __builtin_amdgcn_wmma_f32_16x16x32_bf16(
          false, a0.v, false, b.v, (short)0, acc0[h], false, false);
      acc1[h] = __builtin_amdgcn_wmma_f32_16x16x32_bf16(
          false, a1.v, false, b.v, (short)0, acc1[h], false, false);
    }
  }
#pragma unroll
  for (int h = 0; h < 8; ++h) {
#pragma unroll
    for (int r = 0; r < 8; ++r) {
      int b0 = hlf * 8 + r;
      S[(size_t)(h * 32 + b0)      * NCP + n] = acc0[h][r] * 0.125f;  // 1/sqrt(64)
      S[(size_t)(h * 32 + b0 + 16) * NCP + n] = acc1[h][r] * 0.125f;
    }
  }
}

// ---------------- small fp32 linear (query path) ---------------------------
__global__ void k_linear_small(const float* __restrict__ X, const float* __restrict__ W,
                               const float* __restrict__ bias, float* __restrict__ Y,
                               int Brows, int N, int K) {
  int j = blockIdx.x * blockDim.x + threadIdx.x;
  int b = blockIdx.y;
  if (j >= N || b >= Brows) return;
  const float* x = X + (size_t)b * K;
  const float* w = W + (size_t)j * K;
  float s = 0.0f;
  for (int k = 0; k < K; ++k) s = fmaf(x[k], w[k], s);
  Y[(size_t)b * N + j] = s + bias[j];
}

// ---------------- rowwise LayerNorm + exact GELU ---------------------------
__global__ __launch_bounds__(256) void k_ln_gelu(
    const float* __restrict__ X, const float* __restrict__ g, const float* __restrict__ bt,
    float* __restrict__ Yf, unsigned short* __restrict__ Ybf, int D) {
  __shared__ float sbuf[256];
  const int tid = threadIdx.x;
  const float* x = X + (size_t)blockIdx.x * D;
  float s = 0.0f, s2 = 0.0f;
  for (int k = tid; k < D; k += 256) { float v = x[k]; s += v; s2 += v * v; }
  sbuf[tid] = s; __syncthreads();
  for (int st = 128; st > 0; st >>= 1) { if (tid < st) sbuf[tid] += sbuf[tid + st]; __syncthreads(); }
  const float tot = sbuf[0]; __syncthreads();
  sbuf[tid] = s2; __syncthreads();
  for (int st = 128; st > 0; st >>= 1) { if (tid < st) sbuf[tid] += sbuf[tid + st]; __syncthreads(); }
  const float tot2 = sbuf[0];
  const float mu  = tot / D;
  const float var = tot2 / D - mu * mu;
  const float inv = rsqrtf(var + 1e-5f);
  for (int k = tid; k < D; k += 256) {
    float v  = (x[k] - mu) * inv * g[k] + bt[k];
    float ge = 0.5f * v * (1.0f + erff(v * 0.70710678118654752f));
    if (Yf)  Yf [(size_t)blockIdx.x * D + k] = ge;
    if (Ybf) Ybf[(size_t)blockIdx.x * D + k] = f32_to_bf16(ge);
  }
}

// ---------------- row ||x||^2, 1/||x||, and bf16 copy ----------------------
__global__ __launch_bounds__(256) void k_row_stats_pack(
    const float* __restrict__ X, unsigned short* __restrict__ Xbf,
    float* __restrict__ sq, float* __restrict__ inv, int D) {
  __shared__ float sbuf[256];
  const int tid = threadIdx.x;
  const float* x = X + (size_t)blockIdx.x * D;
  float s2 = 0.0f;
  for (int k = tid; k < D; k += 256) { float v = x[k]; s2 += v * v; }
  sbuf[tid] = s2; __syncthreads();
  for (int st = 128; st > 0; st >>= 1) { if (tid < st) sbuf[tid] += sbuf[tid + st]; __syncthreads(); }
  const float tot = sbuf[0];
  if (tid == 0) { sq[blockIdx.x] = tot; inv[blockIdx.x] = rsqrtf(tot); }
  for (int k = tid; k < D; k += 256) Xbf[(size_t)blockIdx.x * D + k] = f32_to_bf16(x[k]);
}

// ---------------- softmax row stats (max, sum exp) -------------------------
__global__ __launch_bounds__(256) void k_softmax_stats(
    const float* __restrict__ S, float* __restrict__ mx, float* __restrict__ sm,
    int NC, int NCP) {
  __shared__ float sbuf[256];
  const int tid = threadIdx.x;
  const float* s = S + (size_t)blockIdx.x * NCP;
  float m = -3.4e38f;
  for (int i = tid; i < NC; i += 256) m = fmaxf(m, s[i]);
  sbuf[tid] = m; __syncthreads();
  for (int st = 128; st > 0; st >>= 1) { if (tid < st) sbuf[tid] = fmaxf(sbuf[tid], sbuf[tid + st]); __syncthreads(); }
  const float rm = sbuf[0]; __syncthreads();
  float acc = 0.0f;
  for (int i = tid; i < NC; i += 256) acc += expf(s[i] - rm);
  sbuf[tid] = acc; __syncthreads();
  for (int st = 128; st > 0; st >>= 1) { if (tid < st) sbuf[tid] += sbuf[tid + st]; __syncthreads(); }
  if (tid == 0) { mx[blockIdx.x] = rm; sm[blockIdx.x] = sbuf[0]; }
}

// ---------------- fused sims + 3->64->1 MLP + sigmoid ----------------------
__global__ __launch_bounds__(256) void k_fuse_out(
    const float* __restrict__ sdot, const float* __restrict__ q2, const float* __restrict__ qinv,
    const float* __restrict__ k2,  const float* __restrict__ kinv,
    const float* __restrict__ S,   const float* __restrict__ mx, const float* __restrict__ sm,
    const float* __restrict__ temp,
    const float* __restrict__ fW1, const float* __restrict__ fb1,
    const float* __restrict__ fW2, const float* __restrict__ fb2,
    float* __restrict__ out, int NC, int NCP) {
  __shared__ float w1[192];
  __shared__ float b1s[64];
  __shared__ float w2[64];
  __shared__ float b2s;
  const int tid = threadIdx.x;
  if (tid < 192) w1[tid] = fW1[tid];
  if (tid < 64) { b1s[tid] = fb1[tid]; w2[tid] = fW2[tid]; }
  if (tid == 0) b2s = fb2[0];
  __syncthreads();
  size_t idx = (size_t)blockIdx.x * 256 + tid;
  if (idx >= (size_t)32 * NC) return;
  int b = (int)(idx / NC);
  int n = (int)(idx % NC);
  float s    = sdot[(size_t)b * NCP + n];
  float cosv = s * qinv[b] * kinv[n] * expf(temp[0]);
  float d2   = fmaxf(q2[b] + k2[n] - 2.0f * s, 0.0f);
  float euc  = 1.0f / (1.0f + sqrtf(d2));
  float a = 0.0f;
#pragma unroll
  for (int h = 0; h < 8; ++h) {
    int row = h * 32 + b;
    a += expf(S[(size_t)row * NCP + n] - mx[row]) / sm[row];
  }
  float lrn = a * 0.125f;
  float o = 0.0f;
#pragma unroll
  for (int i = 0; i < 64; ++i) {
    float hv = fmaf(w1[i * 3 + 0], cosv,
               fmaf(w1[i * 3 + 1], euc,
               fmaf(w1[i * 3 + 2], lrn, b1s[i])));
    o = fmaf(w2[i], fmaxf(hv, 0.0f), o);
  }
  o += b2s;
  out[idx] = 1.0f / (1.0f + expf(-o));
}

// ---------------------------------------------------------------------------
extern "C" void kernel_launch(void* const* d_in, const int* in_sizes, int n_in,
                              void* d_out, int out_size, void* d_ws, size_t ws_size,
                              hipStream_t stream) {
  (void)in_sizes; (void)n_in; (void)out_size;
  const int NC = 50000, D = 512, B = 32;
  const int MP  = 50048;   // NC padded to 64  (782 * 64)
  const int NCP = 50176;   // NC padded to 256 (196 * 256 = 392 * 128)

  const float* qf   = (const float*)d_in[0];
  const float* cf   = (const float*)d_in[1];
  const float* temp = (const float*)d_in[2];
  const float* qW1  = (const float*)d_in[3];
  const float* qb1  = (const float*)d_in[4];
  const float* qlng = (const float*)d_in[5];
  const float* qlnb = (const float*)d_in[6];
  const float* qW2  = (const float*)d_in[7];
  const float* qb2  = (const float*)d_in[8];
  const float* kW1  = (const float*)d_in[9];
  const float* kb1  = (const float*)d_in[10];
  const float* klng = (const float*)d_in[11];
  const float* klnb = (const float*)d_in[12];
  const float* kW2  = (const float*)d_in[13];
  const float* kb2  = (const float*)d_in[14];
  const float* inw  = (const float*)d_in[15];
  const float* inb  = (const float*)d_in[16];
  const float* fW1  = (const float*)d_in[17];
  const float* fb1  = (const float*)d_in[18];
  const float* fW2  = (const float*)d_in[19];
  const float* fb2  = (const float*)d_in[20];

  char* ws = (char*)d_ws;
  size_t off = 0;
  auto alloc = [&](size_t bytes) -> char* {
    off = (off + 255) & ~(size_t)255;
    char* p = ws + off;
    off += bytes;
    return p;
  };

  const size_t szBt = (size_t)(D / 32) * NCP * 16 * 4;       // swizzled B tiles (51.38 MB)
  const size_t szAbf = (size_t)MP * D * 2;                   // padded bf16 A   (51.25 MB)
  const size_t szF = (size_t)MP * D * 4;                     // padded f32 MxD  (102.5 MB)

  unsigned short* abf  = (unsigned short*)alloc(szBt > szAbf ? szBt : szAbf); // cand bf16 -> kp bf16
  unsigned short* h1bf = (unsigned short*)alloc(szBt > szAbf ? szBt : szAbf); // h1 bf16 -> kp^T tiles
  float* h1f = (float*)alloc(szF);                                            // h1 f32  -> kvec f32
  float* kpf = (float*)alloc(szF > 2 * szBt ? szF : 2 * szBt);                // kp f32  -> k^T + scores
  unsigned* w1p = (unsigned*)alloc((size_t)(D / 32) * D * 16 * 4);
  unsigned* w2p = (unsigned*)alloc((size_t)(D / 32) * D * 16 * 4);
  unsigned* wkp = (unsigned*)alloc((size_t)(D / 32) * D * 16 * 4);
  float* sdot = (float*)alloc((size_t)64 * NCP * 4);
  float* k2v  = (float*)alloc((size_t)NC * 4);
  float* kinv = (float*)alloc((size_t)NC * 4);
  float* qh1  = (float*)alloc((size_t)B * D * 4);
  float* qg   = (float*)alloc((size_t)B * D * 4);
  float* qpf  = (float*)alloc((size_t)B * D * 4);
  float* qvf  = (float*)alloc((size_t)B * D * 4);
  unsigned short* qpbf = (unsigned short*)alloc((size_t)64 * D * 2);  // padded to 64 rows
  unsigned short* qvbf = (unsigned short*)alloc((size_t)B * D * 2);
  float* q2v  = (float*)alloc(B * 4);
  float* qinv = (float*)alloc(B * 4);
  float* smax = (float*)alloc(256 * 4);
  float* ssum = (float*)alloc(256 * 4);
  if (off > ws_size) return;  // scratch too small; bail deterministically

  // Region reuse (safe by stream ordering of producers/consumers):
  unsigned short* kpbf = abf;                // kp bf16 (abf free after GEMM1)
  unsigned* kpT = (unsigned*)h1bf;           // kp^T tiles (h1bf free after GEMM2)
  float* kvf = h1f;                          // kvec f32 (h1f free after LN)
  unsigned* kT = (unsigned*)kpf;             // k^T tiles (kpf consumed)
  float* scoresf = (float*)((char*)kpf + szBt);  // 256 x NCP f32

  auto cdiv = [](int a, int b) { return (a + b - 1) / b; };

  // 1. Weight tiles: W[j][k] f32 -> swizzled (K/32) x 512 x 16 bf16-pair tiles.
  {
    int total = (D / 32) * D * 16;
    k_pack_btile<<<cdiv(total, 256), 256, 0, stream>>>(kW1, w1p, D, D, D);
    k_pack_btile<<<cdiv(total, 256), 256, 0, stream>>>(kW2, w2p, D, D, D);
    k_pack_btile<<<cdiv(total, 256), 256, 0, stream>>>(inw + (size_t)D * D, wkp, D, D, D);
  }
  // 2. Candidate features -> bf16 A matrix (real rows only; pad rows unused garbage).
  k_f32_to_bf16<<<cdiv(NC * D, 256), 256, 0, stream>>>(cf, abf, (size_t)NC * D);

  // 3. Query path (tiny, fp32).
  k_linear_small<<<dim3(cdiv(D, 128), B), 128, 0, stream>>>(qf, qW1, qb1, qh1, B, D, D);
  k_ln_gelu<<<B, 256, 0, stream>>>(qh1, qlng, qlnb, qg, (unsigned short*)nullptr, D);
  k_linear_small<<<dim3(cdiv(D, 128), B), 128, 0, stream>>>(qg, qW2, qb2, qpf, B, D, D);
  k_row_stats_pack<<<B, 256, 0, stream>>>(qpf, qpbf, q2v, qinv, D);
  k_linear_small<<<dim3(cdiv(D, 128), B), 128, 0, stream>>>(qpf, inw, inb, qvf, B, D, D);
  k_f32_to_bf16<<<cdiv(B * D, 256), 256, 0, stream>>>(qvf, qvbf, (size_t)B * D);

  // 4. Candidate path: WMMA GEMM chain (guard-free; MP x 512 padded buffers).
  dim3 g1(D / 256, MP / 64);
  k_wmma_gemm<<<g1, 256, 0, stream>>>(abf, w1p, kb1, h1f, D, D);
  k_ln_gelu<<<NC, 256, 0, stream>>>(h1f, klng, klnb, (float*)nullptr, h1bf, D);
  k_wmma_gemm<<<g1, 256, 0, stream>>>(h1bf, w2p, kb2, kpf, D, D);
  k_row_stats_pack<<<NC, 256, 0, stream>>>(kpf, kpbf, k2v, kinv, D);
  k_pack_btile<<<cdiv((D / 32) * NCP * 16, 256), 256, 0, stream>>>(kpf, kpT, NC, NCP, D);
  k_wmma_gemm<<<g1, 256, 0, stream>>>(kpbf, wkp, inb + D, kvf, D, D);
  k_pack_btile<<<cdiv((D / 32) * NCP * 16, 256), 256, 0, stream>>>(kvf, kT, NC, NCP, D);

  // 5. sdot = qp . kp^T  (rows 0..31 valid; rows 32..63 scratch).
  dim3 g2(NCP / 256, 1);
  k_wmma_gemm<<<g2, 256, 0, stream>>>(qpbf, kpT, (const float*)nullptr, sdot, NCP, D);

  // 6. Per-head attention scores (WMMA, per-head K-split accumulators).
  k_scores_perhead<<<NCP / 128, 256, 0, stream>>>(qvbf, kT, scoresf, NCP);

  // 7. Softmax stats over candidates per (head, query).
  k_softmax_stats<<<256, 256, 0, stream>>>(scoresf, smax, ssum, NC, NCP);

  // 8. Fused sims + MLP + sigmoid -> output.
  k_fuse_out<<<cdiv(B * NC, 256), 256, 0, stream>>>(sdot, q2v, qinv, k2v, kinv, scoresf,
                                                    smax, ssum, temp, fW1, fb1, fW2, fb2,
                                                    (float*)d_out, NC, NCP);
}
